// InferTree_47656957116506
// MI455X (gfx1250) — matrix operand order repaired and checked by reading, not compiled, not run on hardware
//
#include <hip/hip_runtime.h>

typedef __attribute__((ext_vector_type(16))) __bf16          v16bf;
typedef __attribute__((ext_vector_type(8)))  float           v8f;
typedef __attribute__((ext_vector_type(8)))  unsigned short  u16x8;

#define BATCH 4096
#define DIM   2048
#define NB1   25
#define NB2   40
#define RCOLS 26                      // root logits columns
#define NCOLS 41                      // per-node logits columns
#define LCOLS (RCOLS + NB1 * NCOLS)   // 1051
#define OUTC  (NB1 * NB2)             // 1000
#define KCH   64                      // DIM/32 K-chunks

// ---------- bf16 split helpers (round-to-nearest-even) ----------
__device__ __forceinline__ unsigned short f32_to_bf16_rne(float f) {
  unsigned int u = __builtin_bit_cast(unsigned int, f);
  u += 0x7FFFu + ((u >> 16) & 1u);
  return (unsigned short)(u >> 16);
}
__device__ __forceinline__ float bf16_bits_to_f32(unsigned short h) {
  return __builtin_bit_cast(float, ((unsigned int)h) << 16);
}

union AFrag { v16bf v; u16x8 h[2]; };

// =====================================================================
// Kernel 0a: split x (fp32) -> xh/xl (bf16 bits), row-major. 8 elems/thread.
// =====================================================================
__global__ void __launch_bounds__(256)
cvt_x(const float* __restrict__ x,
      unsigned short* __restrict__ xh, unsigned short* __restrict__ xl)
{
  const size_t u = (size_t)blockIdx.x * 256 + threadIdx.x;   // unit of 8 elements
  const float4* px = (const float4*)x + u * 2;
  const float4 f0 = px[0];
  const float4 f1 = px[1];
  const float f[8] = { f0.x, f0.y, f0.z, f0.w, f1.x, f1.y, f1.z, f1.w };
  u16x8 h, l;
  #pragma unroll
  for (int e = 0; e < 8; ++e) {
    const unsigned short hh = f32_to_bf16_rne(f[e]);
    h[e] = hh;
    l[e] = f32_to_bf16_rne(f[e] - bf16_bits_to_f32(hh));
  }
  *(u16x8*)(xh + u * 8) = h;
  *(u16x8*)(xl + u * 8) = l;
}

// =====================================================================
// Kernel 0b: build pre-swizzled weight fragments.
//   Unit u = ((g*64 + q)*3 + s)*32 + lane ; each unit owns one lane's
//   B fragment for (group g, K-chunk q, subtile s): 16 hi + 16 lo ushorts
//   contiguous (64B).  Fragment element e <-> K = (lane>>4)*8 + e + (e>=8 ? 8 : 0),
//   column c = s*16 + (lane&15) (zero-padded beyond panel width).
// =====================================================================
__global__ void __launch_bounds__(256)
cvt_w(const float* __restrict__ Wroot, const float* __restrict__ Wlvl2,
      unsigned short* __restrict__ wf)
{
  const int u = (int)blockIdx.x * 256 + (int)threadIdx.x;
  const int TOT = 26 * KCH * 3 * 32;
  if (u >= TOT) return;
  const int lane = u & 31;
  const int rest = u >> 5;
  const int s  = rest % 3;
  const int gq = rest / 3;
  const int q  = gq & 63;
  const int g  = gq >> 6;

  const float* Wp; int ncols, stride;
  if (g == 0) { Wp = Wroot; ncols = RCOLS; stride = RCOLS; }
  else        { Wp = Wlvl2 + (size_t)(g - 1) * DIM * NCOLS; ncols = NCOLS; stride = NCOLS; }

  const int half = lane >> 4, l16 = lane & 15;
  const int c = s * 16 + l16;
  const int kbase = q * 32 + half * 8;

  u16x8 h0, h1, l0, l1;
  #pragma unroll
  for (int e = 0; e < 16; ++e) {
    const float wv = (c < ncols) ? Wp[(size_t)(kbase + e + (e >= 8 ? 8 : 0)) * stride + c] : 0.0f;
    const unsigned short hh = f32_to_bf16_rne(wv);
    const unsigned short ll = f32_to_bf16_rne(wv - bf16_bits_to_f32(hh));
    if (e < 8) { h0[e] = hh; l0[e] = ll; }
    else       { h1[e - 8] = hh; l1[e - 8] = ll; }
  }
  unsigned short* dst = wf + (size_t)u * 32;
  *(u16x8*)(dst)      = h0;
  *(u16x8*)(dst + 8)  = h1;
  *(u16x8*)(dst + 16) = l0;
  *(u16x8*)(dst + 24) = l1;
}

// =====================================================================
// Kernel 1: hierarchical GEMM, pure vector-load + WMMA inner loop.
//   grid = (32 M-blocks of 128 rows, 26 groups); block = 128 thr = 4 waves.
//   Each wave: 32 rows (2 A fragments) x 48 cols (3 subtiles), K chunks of 32.
//   bf16x3 split accumulation in fp32 (fp32-grade accuracy).
// =====================================================================
__global__ void __launch_bounds__(128)
hs_gemm_wmma(const unsigned short* __restrict__ xh,
             const unsigned short* __restrict__ xl,
             const unsigned short* __restrict__ wf,
             float* __restrict__ logits)
{
  const int t    = (int)threadIdx.x;
  const int lane = t & 31;
  const int wave = t >> 5;
  const int half = lane >> 4;
  const int l16  = lane & 15;

  const int row0 = (int)blockIdx.x * 128 + wave * 32;   // wave owns 32 rows
  const int g    = (int)blockIdx.y;
  int ncols, obase;
  if (g == 0) { ncols = RCOLS; obase = 0; }
  else        { ncols = NCOLS; obase = RCOLS + (g - 1) * NCOLS; }

  v8f acc[2][3] = {};

  const size_t aoff0 = (size_t)(row0 + l16)      * DIM + half * 8;
  const size_t aoff1 = (size_t)(row0 + 16 + l16) * DIM + half * 8;
  const unsigned short* wbase = wf + (((size_t)g * KCH * 3) * 32 + lane) * 32;

  for (int q = 0; q < KCH; ++q) {
    // ---- A fragments (hi/lo), ISA 16-bit A 16x32 layout: two 16B segs ----
    AFrag ah[2], al[2];
    const size_t k0 = (size_t)q * 32;
    ah[0].h[0] = *(const u16x8*)(xh + aoff0 + k0);
    ah[0].h[1] = *(const u16x8*)(xh + aoff0 + k0 + 16);
    al[0].h[0] = *(const u16x8*)(xl + aoff0 + k0);
    al[0].h[1] = *(const u16x8*)(xl + aoff0 + k0 + 16);
    ah[1].h[0] = *(const u16x8*)(xh + aoff1 + k0);
    ah[1].h[1] = *(const u16x8*)(xh + aoff1 + k0 + 16);
    al[1].h[0] = *(const u16x8*)(xl + aoff1 + k0);
    al[1].h[1] = *(const u16x8*)(xl + aoff1 + k0 + 16);

    // ---- 3 N-subtiles; B fragments are pre-swizzled 64B blobs ----
    #pragma unroll
    for (int s = 0; s < 3; ++s) {
      const unsigned short* wb = wbase + ((size_t)(q * 3 + s) * 32) * 32;
      const v16bf bh = *(const v16bf*)(wb);
      const v16bf bl = *(const v16bf*)(wb + 16);
      #pragma unroll
      for (int m = 0; m < 2; ++m) {
        acc[m][s] = __builtin_amdgcn_wmma_f32_16x16x32_bf16(
            false, ah[m].v, false, bh, (short)0, acc[m][s], false, false);
        acc[m][s] = __builtin_amdgcn_wmma_f32_16x16x32_bf16(
            false, al[m].v, false, bh, (short)0, acc[m][s], false, false);
        acc[m][s] = __builtin_amdgcn_wmma_f32_16x16x32_bf16(
            false, ah[m].v, false, bl, (short)0, acc[m][s], false, false);
      }
    }
  }

  // ---- write C (32-bit C/D 16x16 layout: VGPR j -> M = j + half*8) ----
  #pragma unroll
  for (int m = 0; m < 2; ++m) {
    const int mbase = row0 + m * 16 + half * 8;
    #pragma unroll
    for (int j = 0; j < 8; ++j) {
      #pragma unroll
      for (int s = 0; s < 3; ++s) {
        const int cl = s * 16 + l16;
        if (cl < ncols)
          logits[(size_t)(mbase + j) * LCOLS + obase + cl] = acc[m][s][j];
      }
    }
  }
}

// =====================================================================
// Kernel 2: per-row log-softmaxes, output combine, per-row penalty part.
// =====================================================================
__global__ void __launch_bounds__(32)
hs_softmax(const float* __restrict__ logits,
           const int*   __restrict__ labels,
           const float* __restrict__ broot,
           const float* __restrict__ blvl2,
           float* __restrict__ out,
           float* __restrict__ pen_partial)
{
  const int b = (int)blockIdx.x;
  const int t = (int)threadIdx.x;
  __shared__ float root_lp[RCOLS];
  __shared__ float pen[32];

  const float* lrow = logits + (size_t)b * LCOLS;

  if (t == 0) {
    float mx = -3.0e38f;
    for (int c = 0; c < RCOLS; ++c) mx = fmaxf(mx, lrow[c] + broot[c]);
    float s = 0.0f;
    for (int c = 0; c < RCOLS; ++c) s += expf(lrow[c] + broot[c] - mx);
    const float lse = mx + logf(s);
    for (int c = 0; c < RCOLS; ++c) root_lp[c] = lrow[c] + broot[c] - lse;
  }
  __syncthreads();

  const int lab = labels[b];
  const int j = lab / NB2;
  const int i = lab - j * NB2;

  float p = 0.0f;
  if (t < NB1) {
    const int n = t;
    const float* l2 = lrow + RCOLS + n * NCOLS;
    const float* bb = blvl2 + n * NCOLS;
    float mx = -3.0e38f;
    for (int c = 0; c < NCOLS; ++c) mx = fmaxf(mx, l2[c] + bb[c]);
    float s = 0.0f;
    for (int c = 0; c < NCOLS; ++c) s += expf(l2[c] + bb[c] - mx);
    const float lse = mx + logf(s);
    const float rp = expf(root_lp[n + 1]);
    float* orow = out + (size_t)b * OUTC + n * NB2;
    for (int c = 0; c < NB2; ++c)
      orow[c] = expf(l2[c + 1] + bb[c + 1] - lse) * rp;
    const int sub = (j == n) ? (i + 1) : 0;
    p = 0.5f * (lse - (l2[sub] + bb[sub]));   // 0.5 * (-lvl2_lp[n][sub])
  }
  pen[t] = p;
  __syncthreads();

  if (t == 0) {
    float s = -root_lp[j + 1];
    for (int k = 0; k < NB1; ++k) s += pen[k];
    pen_partial[b] = s;
  }
}

// =====================================================================
// Kernel 3: deterministic tree reduction of penalty partials.
// =====================================================================
__global__ void __launch_bounds__(256)
hs_reduce(const float* __restrict__ pen_partial, float* __restrict__ out_pen)
{
  __shared__ float sm[256];
  const int t = (int)threadIdx.x;
  float a = 0.0f;
  for (int idx = t; idx < BATCH; idx += 256) a += pen_partial[idx];
  sm[t] = a;
  __syncthreads();
  for (int w = 128; w > 0; w >>= 1) {
    if (t < w) sm[t] += sm[t + w];
    __syncthreads();
  }
  if (t == 0) out_pen[0] = sm[0] / (float)BATCH;
}

// =====================================================================
extern "C" void kernel_launch(void* const* d_in, const int* in_sizes, int n_in,
                              void* d_out, int out_size, void* d_ws, size_t ws_size,
                              hipStream_t stream) {
  (void)in_sizes; (void)n_in; (void)out_size; (void)ws_size;
  const float* x      = (const float*)d_in[0];
  const int*   labels = (const int*)d_in[1];
  const float* Wroot  = (const float*)d_in[2];
  const float* broot  = (const float*)d_in[3];
  const float* Wlvl2  = (const float*)d_in[4];
  const float* blvl2  = (const float*)d_in[5];

  float* out     = (float*)d_out;

  // workspace layout
  float*          logits  = (float*)d_ws;                         // BATCH*LCOLS f32
  float*          penpart = logits + (size_t)BATCH * LCOLS;       // BATCH f32
  unsigned short* xh      = (unsigned short*)(penpart + BATCH);   // BATCH*DIM u16
  unsigned short* xl      = xh + (size_t)BATCH * DIM;             // BATCH*DIM u16
  unsigned short* wf      = xl + (size_t)BATCH * DIM;             // 26*64*3*32*32 u16

  // 0) one-shot precision split (memory-bound, ~100MB traffic)
  cvt_x<<<(BATCH * DIM / 8 + 255) / 256, 256, 0, stream>>>(x, xh, xl);
  cvt_w<<<(26 * KCH * 3 * 32 + 255) / 256, 256, 0, stream>>>(Wroot, Wlvl2, wf);

  // 1) WMMA GEMM: logits = x @ [W_root | W_lvl2[0..24]]
  dim3 ggrid(BATCH / 128, NB1 + 1);
  hs_gemm_wmma<<<ggrid, 128, 0, stream>>>(xh, xl, wf, logits);

  // 2) softmaxes + combine + per-row penalty; 3) reduce
  hs_softmax<<<BATCH, 32, 0, stream>>>(logits, labels, broot, blvl2, out, penpart);
  hs_reduce<<<1, 256, 0, stream>>>(penpart, out + (size_t)BATCH * OUTC);
}
